// WordLevelLayer_24627342475657
// MI455X (gfx1250) — matrix operand order repaired and checked by reading, compiled, not run on hardware
//
#include <hip/hip_runtime.h>

// ---------------------------------------------------------------------------
// WordLevelLayer on MI455X (gfx1250, wave32, WMMA)
//
// Sequential topological scans -> latency-bound; whole problem is L2-resident
// (192MB L2 >> 11MB working set).  One persistent 4-wave workgroup per phase.
// Per-node matvec done with v_wmma_f32_16x16x32_bf16:
//   forward:  y = concat(m0*x0, m1*x1)[1x256] @ [W_f[e0]; W_f[e1]][256x128]
//   backward: y_k = x[1x128] @ W_b[e_k][128x128]   (two chains, 4 waves)
// Weights are pre-converted once to bf16 WMMA B-operand fragments in d_ws.
// ---------------------------------------------------------------------------

#define D   128
#define NTI 8      // 16-wide N tiles per 128 outputs
#define KTI 4      // 32-wide K tiles per 128 inputs

typedef __attribute__((ext_vector_type(16))) __bf16 v16bf;
typedef __attribute__((ext_vector_type(8)))  float  v8f;

// ---------------------------------------------------------------------------
// Kernel 0: copy feat -> wh (d_out), float4 grid-stride
// ---------------------------------------------------------------------------
__global__ void wl_copy_feat(const float4* __restrict__ src,
                             float4* __restrict__ dst, long n4) {
  long stride = (long)gridDim.x * blockDim.x;
  for (long i = (long)blockIdx.x * blockDim.x + threadIdx.x; i < n4; i += stride)
    dst[i] = src[i];
}

// ---------------------------------------------------------------------------
// Kernel 1: fp32 W_f / W_b -> bf16 WMMA B-fragments in workspace.
// Fragment layout per (etype e, ktile kt, ntile nt):
//   32 lanes x 16 bf16 elements; lane l covers column n = nt*16 + (l&15),
//   K-half (l>>4)*16; element pair (2v,2v+1) holds K = kt*32 + (l>>4)*16 + 2v+{0,1}.
// Linear frag-element index m = (((e*4+kt)*8+nt)*32+lane)*16+elem.
// ---------------------------------------------------------------------------
__global__ void wl_prep_weights(const float* __restrict__ Wf,
                                const float* __restrict__ Wb,
                                __bf16* __restrict__ ws, long perM) {
  long stride = (long)gridDim.x * blockDim.x;
  for (long t = (long)blockIdx.x * blockDim.x + threadIdx.x; t < 2 * perM; t += stride) {
    long m = t;
    const float* W = Wf;
    long base = 0;
    if (m >= perM) { m -= perM; W = Wb; base = perM; }
    int elem = (int)(m & 15);
    int lane = (int)((m >> 4) & 31);
    int nt   = (int)((m >> 9) & 7);
    int kt   = (int)((m >> 12) & 3);
    int e    = (int)(m >> 14);
    int n = nt * 16 + (lane & 15);
    int k = kt * 32 + (lane >> 4) * 16 + (elem >> 1) * 2 + (elem & 1);
    ws[base + m] = (__bf16)W[(long)e * D * D + (long)k * D + n];
  }
}

// ---------------------------------------------------------------------------
// Build one 16x32 bf16 A-fragment row (M=0 only) from a 32-float slice.
// A layout: lanes 0-15 hold M=lane; element pair (2v,2v+1) -> K=(l>>4)*16 + 2v+{0,1}
// (matches B fragments).  Only lanes with (lane&15)==0 carry data.
// ---------------------------------------------------------------------------
__device__ __forceinline__ v16bf make_afrag(const float* __restrict__ xe,
                                            int khalf, bool mrow) {
  v16bf a;
#pragma unroll
  for (int v = 0; v < 8; ++v) {
    int kl = khalf * 16 + (v << 1);
    a[2 * v]     = mrow ? (__bf16)xe[kl]     : (__bf16)0.0f;
    a[2 * v + 1] = mrow ? (__bf16)xe[kl + 1] : (__bf16)0.0f;
  }
  return a;
}

// ---------------------------------------------------------------------------
// Kernel 2: forward topological scan.  One workgroup, 4 waves; wave w owns
// N-tiles {2w, 2w+1}.  K=256 (two edges stacked) -> 8 WMMA per tile chain.
// ---------------------------------------------------------------------------
__global__ void __launch_bounds__(128)
wl_fwd_scan(float* __restrict__ wh,
            const __bf16* __restrict__ wfrag,
            const float* __restrict__ bf,
            const int* __restrict__ in_src,
            const int* __restrict__ in_etype,
            const float* __restrict__ in_mask,
            int first_prop, int P) {
  __shared__ float xbuf[2 * D];
  const int lane  = threadIdx.x & 31;
  const int wave  = threadIdx.x >> 5;
  const int khalf = lane >> 4;
  const bool mrow = (lane & 15) == 0;
  const int nt0 = wave * 2, nt1 = wave * 2 + 1;

  for (int r = 0; r < P; ++r) {
    if (r + 1 < P) {  // global_prefetch_b8 next step's metadata
      __builtin_prefetch(in_src + 2 * (r + 1), 0, 3);
      __builtin_prefetch(in_etype + 2 * (r + 1), 0, 3);
    }
    const int   node = first_prop + r;
    const int   s0 = in_src[2 * r],    s1 = in_src[2 * r + 1];
    const int   e0 = in_etype[2 * r],  e1 = in_etype[2 * r + 1];
    const float m0 = in_mask[2 * r],   m1 = in_mask[2 * r + 1];
    const float rinv = 1.0f / (m0 + m1);

    if (wave == 0) {  // gather + mask-scale both source vectors into LDS
#pragma unroll
      for (int j = 0; j < 4; ++j) {
        int i = lane + 32 * j;
        xbuf[i]     = m0 * wh[(long)s0 * D + i];
        xbuf[D + i] = m1 * wh[(long)s1 * D + i];
      }
    }
    __syncthreads();

    v16bf a[8];
#pragma unroll
    for (int kt = 0; kt < 8; ++kt)
      a[kt] = make_afrag(&xbuf[(kt >> 2) * D + (kt & 3) * 32], khalf, mrow);

    v8f c0 = {}; v8f c1 = {};
#pragma unroll
    for (int kt = 0; kt < 8; ++kt) {
      const int  e  = (kt < 4) ? e0 : e1;
      const long fb = ((long)e * KTI + (kt & 3)) * NTI;
      v16bf b0 = *(const v16bf*)(wfrag + ((fb + nt0) * 32 + lane) * 16);
      v16bf b1 = *(const v16bf*)(wfrag + ((fb + nt1) * 32 + lane) * 16);
      c0 = __builtin_amdgcn_wmma_f32_16x16x32_bf16(false, a[kt], false, b0,
                                                   (short)0, c0, false, false);
      c1 = __builtin_amdgcn_wmma_f32_16x16x32_bf16(false, a[kt], false, b1,
                                                   (short)0, c1, false, false);
    }

    if (lane < 16) {  // row M=0 lives in VGPR0 of lanes 0-15 (N = lane)
      int n0 = nt0 * 16 + lane, n1 = nt1 * 16 + lane;
      float y0 = (c0[0] + m0 * bf[e0 * D + n0] + m1 * bf[e1 * D + n0]) * rinv;
      float y1 = (c1[0] + m0 * bf[e0 * D + n1] + m1 * bf[e1 * D + n1]) * rinv;
      wh[(long)node * D + n0] = fmaxf(y0, 0.0f);
      wh[(long)node * D + n1] = fmaxf(y1, 0.0f);
    }
    __threadfence_block();   // make step-r stores visible to step-(r+1) gathers
    __syncthreads();
  }
}

// ---------------------------------------------------------------------------
// Kernel 3: backward scan (descending r).  Waves 0,1 -> out-edge 0 (stored
// first), waves 2,3 -> out-edge 1 (stored second => wins on duplicate dst,
// matching the reference's sequential k-loop).  Each wave: 4 N-tiles, K=128.
// ---------------------------------------------------------------------------
__global__ void __launch_bounds__(128)
wl_bwd_scan(float* __restrict__ wh,
            const __bf16* __restrict__ wfrag,
            const float* __restrict__ bb,
            const int* __restrict__ out_dst,
            const int* __restrict__ out_etype,
            const float* __restrict__ out_mask,
            int first_prop, int P) {
  __shared__ float xbuf[D];
  const int lane  = threadIdx.x & 31;
  const int wave  = threadIdx.x >> 5;
  const int khalf = lane >> 4;
  const bool mrow = (lane & 15) == 0;
  const int edge = wave >> 1;
  const int ntb  = (wave & 1) * 4;

  for (int r = P - 1; r >= 0; --r) {
    if (r > 0) {
      __builtin_prefetch(out_dst + 2 * (r - 1), 0, 3);
      __builtin_prefetch(out_etype + 2 * (r - 1), 0, 3);
    }
    const int   node = first_prop + r;
    const int   dst = out_dst[2 * r + edge];
    const int   e   = out_etype[2 * r + edge];
    const float m   = out_mask[2 * r + edge];

    if (wave == 0) {
#pragma unroll
      for (int j = 0; j < 4; ++j) {
        int i = lane + 32 * j;
        xbuf[i] = wh[(long)node * D + i];
      }
    }
    __syncthreads();

    v16bf a[4];
#pragma unroll
    for (int kt = 0; kt < 4; ++kt)
      a[kt] = make_afrag(&xbuf[kt * 32], khalf, mrow);

    v8f c[4] = {};
#pragma unroll
    for (int kt = 0; kt < 4; ++kt) {
      const long fb = ((long)e * KTI + kt) * NTI;
#pragma unroll
      for (int j = 0; j < 4; ++j) {
        v16bf b = *(const v16bf*)(wfrag + ((fb + ntb + j) * 32 + lane) * 16);
        c[j] = __builtin_amdgcn_wmma_f32_16x16x32_bf16(false, a[kt], false, b,
                                                       (short)0, c[j], false, false);
      }
    }

    __syncthreads();  // all xbuf reads done; begin ordered store phases
    if (edge == 0 && m > 0.0f && lane < 16) {
#pragma unroll
      for (int j = 0; j < 4; ++j) {
        int n = (ntb + j) * 16 + lane;
        wh[(long)dst * D + n] = fmaxf(c[j][0] + bb[e * D + n], 0.0f);
      }
    }
    __threadfence_block();
    __syncthreads();
    if (edge == 1 && m > 0.0f && lane < 16) {
#pragma unroll
      for (int j = 0; j < 4; ++j) {
        int n = (ntb + j) * 16 + lane;
        wh[(long)dst * D + n] = fmaxf(c[j][0] + bb[e * D + n], 0.0f);
      }
    }
    __threadfence_block();
    __syncthreads();
  }
}

// ---------------------------------------------------------------------------
extern "C" void kernel_launch(void* const* d_in, const int* in_sizes, int n_in,
                              void* d_out, int out_size, void* d_ws, size_t ws_size,
                              hipStream_t stream) {
  const float* feat      = (const float*)d_in[0];
  const float* W_f       = (const float*)d_in[1];
  const float* b_f       = (const float*)d_in[2];
  const float* W_b       = (const float*)d_in[3];
  const float* b_b       = (const float*)d_in[4];
  const int*   in_src    = (const int*)d_in[5];
  const int*   in_etype  = (const int*)d_in[6];
  const float* in_mask   = (const float*)d_in[7];
  const int*   out_dst   = (const int*)d_in[8];
  const int*   out_etype = (const int*)d_in[9];
  const float* out_mask  = (const float*)d_in[10];
  float* wh = (float*)d_out;

  const int  E          = in_sizes[1] / (D * D);
  const int  P          = in_sizes[5] / 2;
  const int  n_nodes    = in_sizes[0] / D;
  const int  first_prop = n_nodes - P;
  const long perM       = (long)E * D * D;   // bf16 frag elements per matrix

  __bf16* wsW = (__bf16*)d_ws;

  wl_copy_feat<<<512, 256, 0, stream>>>((const float4*)feat, (float4*)wh,
                                        (long)in_sizes[0] / 4);
  {
    long total = 2 * perM;
    int blocks = (int)((total + 255) / 256);
    if (blocks > 2048) blocks = 2048;
    wl_prep_weights<<<blocks, 256, 0, stream>>>(W_f, W_b, wsW, perM);
  }
  wl_fwd_scan<<<1, 128, 0, stream>>>(wh, wsW, b_f, in_src, in_etype, in_mask,
                                     first_prop, P);
  wl_bwd_scan<<<1, 128, 0, stream>>>(wh, wsW + perM, b_b, out_dst, out_etype,
                                     out_mask, first_prop, P);
}
